// Attention_Sep_7645041787037
// MI455X (gfx1250) — matrix operand order repaired and checked by reading, not compiled, not run on hardware
//
#include <hip/hip_runtime.h>

typedef __attribute__((ext_vector_type(16))) __bf16 v16bf;
typedef __attribute__((ext_vector_type(8)))  __bf16 v8bf;
typedef __attribute__((ext_vector_type(4)))  __bf16 v4bf;
typedef __attribute__((ext_vector_type(8)))  float  v8f;

namespace {
constexpr int kB = 16, kN = 1025, kC = 768, kH = 12, kNB = 4, kD = 64, kSPB = 256, kC3 = 2304;
constexpr int kM = kB * kN;  // 16400 rows
constexpr float kScale = 0.125f;  // 64^-0.5
}

static __device__ __forceinline__ v8f wmma_bf16(v16bf a, v16bf b, v8f c) {
  return __builtin_amdgcn_wmma_f32_16x16x32_bf16(false, a, false, b, (short)0, c, false, false);
}

// Assemble a 16-element bf16 fragment from two contiguous 8-element (16B) chunks.
static __device__ __forceinline__ v16bf make_frag(const __bf16* p0, const __bf16* p1) {
  v8bf a = *(const v8bf*)p0;
  v8bf b = *(const v8bf*)p1;
  v16bf r;
#pragma unroll
  for (int j = 0; j < 8; ++j) { r[j] = a[j]; r[j + 8] = b[j]; }
  return r;
}

// CDNA5 async global->LDS copy, 8 bytes per lane. Tracked on ASYNCcnt.
// Generic pointers to __shared__ carry the LDS byte offset in their low 32 bits
// (flat-aperture rule).
static __device__ __forceinline__ void async_copy_b64(void* lds_dst, const void* gsrc) {
  asm volatile("global_load_async_to_lds_b64 %0, %1, off"
               :
               : "v"((unsigned)(unsigned long long)lds_dst),
                 "v"((unsigned long long)gsrc)
               : "memory");
}
static __device__ __forceinline__ void wait_asynccnt0() {
  asm volatile("s_wait_asynccnt 0x0" ::: "memory");
}

// ---------------------------------------------------------------------------
// fp32 -> bf16 conversion
// ---------------------------------------------------------------------------
__global__ void cvt_f32_bf16_kernel(const float* __restrict__ in, __bf16* __restrict__ out, long n) {
  long i = (long)blockIdx.x * blockDim.x + threadIdx.x;
  if (i < n) out[i] = (__bf16)in[i];
}

// ---------------------------------------------------------------------------
// bf16 WMMA GEMM: C[M,Nn] = A[M,K] * B[K,Nn]  (+bias, f32 out) or bf16 out.
// Tile 128x128x32, 256 threads = 8 waves (2x4), each wave 64x32 (8 WMMA frags).
// Double-buffered LDS: global loads for step i+1 overlap the 8 WMMAs of step i;
// one barrier per K-step.
// ---------------------------------------------------------------------------
template <bool OUT_BF16>
__global__ __launch_bounds__(256) void gemm_wmma_kernel(
    const __bf16* __restrict__ A, const __bf16* __restrict__ Bm,
    const float* __restrict__ bias, void* __restrict__ Cout,
    int M, int Nn, int K) {
  __shared__ __bf16 Ash[2][128 * 32];  // [buf][row][k]
  __shared__ __bf16 Bt[2][128 * 32];   // [buf][col][k] (B staged transposed)
  const int tid = threadIdx.x;
  const int lane = tid & 31, wave = tid >> 5;
  const int lrow = lane & 15, lhalf = lane >> 4;
  const int wm = wave >> 2, wn = wave & 3;
  const int bm = blockIdx.x * 128, bn = blockIdx.y * 128;

  // staging roles (fixed per thread)
  const int ar = tid >> 1;              // A row 0..127
  const int ah = (tid & 1) * 16;        // A k-chunk 0/16
  const int bkk = tid & 31;             // B k 0..31
  const int bn0 = (tid >> 5) * 16;      // B col group

  v8f acc[4][2];
#pragma unroll
  for (int mi = 0; mi < 4; ++mi)
#pragma unroll
    for (int ni = 0; ni < 2; ++ni)
#pragma unroll
      for (int v = 0; v < 8; ++v) acc[mi][ni][v] = 0.0f;

  auto load_glb = [&](int k0, v8bf& a0, v8bf& a1, v8bf& b0, v8bf& b1) {
#pragma unroll
    for (int j = 0; j < 8; ++j) { a0[j] = (__bf16)0.f; a1[j] = (__bf16)0.f; }
    const int gr = bm + ar;
    if (gr < M) {
      const __bf16* src = A + (size_t)gr * K + k0 + ah;
      a0 = *(const v8bf*)(src);
      a1 = *(const v8bf*)(src + 8);
    }
    const __bf16* bsrc = Bm + (size_t)(k0 + bkk) * Nn + bn + bn0;
    b0 = *(const v8bf*)(bsrc);
    b1 = *(const v8bf*)(bsrc + 8);
  };
  auto store_lds = [&](int buf, v8bf a0, v8bf a1, v8bf b0, v8bf b1) {
    *(v8bf*)(&Ash[buf][ar * 32 + ah]) = a0;
    *(v8bf*)(&Ash[buf][ar * 32 + ah + 8]) = a1;
#pragma unroll
    for (int j = 0; j < 8; ++j) {
      Bt[buf][(bn0 + j) * 32 + bkk] = b0[j];
      Bt[buf][(bn0 + 8 + j) * 32 + bkk] = b1[j];
    }
  };

  const int steps = K >> 5;
  {
    v8bf a0, a1, b0, b1;
    load_glb(0, a0, a1, b0, b1);
    store_lds(0, a0, a1, b0, b1);
  }
  __syncthreads();

  for (int i = 0; i < steps; ++i) {
    const int cur = i & 1;
    v8bf na0, na1, nb0, nb1;
    const bool has_next = (i + 1 < steps);
    if (has_next) load_glb((i + 1) * 32, na0, na1, nb0, nb1);  // overlaps WMMAs

#pragma unroll
    for (int mi = 0; mi < 4; ++mi) {
      const __bf16* ap = &Ash[cur][(wm * 64 + mi * 16 + lrow) * 32];
      v16bf afrag = make_frag(ap + lhalf * 8, ap + 16 + lhalf * 8);
#pragma unroll
      for (int ni = 0; ni < 2; ++ni) {
        const __bf16* bp = &Bt[cur][(wn * 32 + ni * 16 + lrow) * 32 + lhalf * 8];
        v16bf bfrag = make_frag(bp, bp + 16);
        acc[mi][ni] = wmma_bf16(afrag, bfrag, acc[mi][ni]);
      }
    }

    if (has_next) {
      store_lds(cur ^ 1, na0, na1, nb0, nb1);
      __syncthreads();
    }
  }

#pragma unroll
  for (int mi = 0; mi < 4; ++mi)
#pragma unroll
    for (int ni = 0; ni < 2; ++ni)
#pragma unroll
      for (int v = 0; v < 8; ++v) {
        const int r = bm + wm * 64 + mi * 16 + lhalf * 8 + v;
        const int c = bn + wn * 32 + ni * 16 + lrow;
        if (r < M) {
          if (OUT_BF16) {
            ((__bf16*)Cout)[(size_t)r * Nn + c] = (__bf16)acc[mi][ni][v];
          } else {
            ((float*)Cout)[(size_t)r * Nn + c] = acc[mi][ni][v] + bias[c];
          }
        }
      }
}

// ---------------------------------------------------------------------------
// cls global attention: one block per (b,h). Two-pass softmax over 1025 keys.
// cls = attn(q0, K, V) + x[:,0]; writes cls (f32) and y row 0 (bf16).
// ---------------------------------------------------------------------------
__global__ __launch_bounds__(256) void cls_attn_kernel(
    const __bf16* __restrict__ qkv, const float* __restrict__ x,
    float* __restrict__ cls_f32, __bf16* __restrict__ y) {
  const int b = blockIdx.x / kH;
  const int h = blockIdx.x % kH;
  __shared__ float q0[kD];
  __shared__ float logits[kN];
  __shared__ float red[256];
  const int tid = threadIdx.x;
  const __bf16* base = qkv + (size_t)b * kN * kC3;

  if (tid < kD) q0[tid] = (float)base[h * kD + tid];
  __syncthreads();

  for (int j = tid; j < kN; j += 256) {
    const __bf16* krow = base + (size_t)j * kC3 + kC + h * kD;
    float s = 0.f;
#pragma unroll 8
    for (int dd = 0; dd < kD; ++dd) s += q0[dd] * (float)krow[dd];
    logits[j] = s * kScale;
  }
  __syncthreads();

  float mx = -3.0e38f;
  for (int j = tid; j < kN; j += 256) mx = fmaxf(mx, logits[j]);
  red[tid] = mx;
  __syncthreads();
  for (int s2 = 128; s2 > 0; s2 >>= 1) {
    if (tid < s2) red[tid] = fmaxf(red[tid], red[tid + s2]);
    __syncthreads();
  }
  mx = red[0];
  __syncthreads();

  float sm = 0.f;
  for (int j = tid; j < kN; j += 256) {
    float e = __expf(logits[j] - mx);
    logits[j] = e;
    sm += e;
  }
  red[tid] = sm;
  __syncthreads();
  for (int s2 = 128; s2 > 0; s2 >>= 1) {
    if (tid < s2) red[tid] += red[tid + s2];
    __syncthreads();
  }
  sm = red[0];
  __syncthreads();

  const int dim = tid & 63, chunk = tid >> 6;
  float o = 0.f;
  for (int j = chunk; j < kN; j += 4)
    o += logits[j] * (float)base[(size_t)j * kC3 + 2 * kC + h * kD + dim];
  red[tid] = o;
  __syncthreads();
  if (chunk == 0) {
    o = red[dim] + red[64 + dim] + red[128 + dim] + red[192 + dim];
    const int c = h * kD + dim;
    const float val = o / sm + x[(size_t)b * kN * kC + c];
    cls_f32[b * kC + c] = val;
    y[(size_t)b * kN * kC + c] = (__bf16)val;
  }
}

// ---------------------------------------------------------------------------
// qkv_c = cls @ W_qkv  (tiny GEMV: 16 x 2304 outputs, K=768)
// ---------------------------------------------------------------------------
__global__ __launch_bounds__(256) void qkvc_kernel(
    const float* __restrict__ cls_f32, const __bf16* __restrict__ Wqkv,
    __bf16* __restrict__ qkvc) {
  const int b = blockIdx.x;
  const int col = blockIdx.y * 256 + threadIdx.x;
  __shared__ float cl[kC];
  for (int i = threadIdx.x; i < kC; i += 256) cl[i] = cls_f32[b * kC + i];
  __syncthreads();
  float s = 0.f;
#pragma unroll 4
  for (int k = 0; k < kC; ++k) s += cl[k] * (float)Wqkv[(size_t)k * kC3 + col];
  qkvc[(size_t)b * kC3 + col] = (__bf16)s;
}

// ---------------------------------------------------------------------------
// Branch attention: one block per (br, h, b). 512 threads = 16 waves; wave w
// owns query rows w*16..w*16+15 (tokens 1+br*256+...). Flash-style over 8
// key-blocks of 32 keys; the shared cls key/value (from qkv_c) is folded in
// as the initial softmax state. K tiles staged with CDNA5 async-to-LDS.
// ---------------------------------------------------------------------------
__global__ __launch_bounds__(512) void branch_attn_kernel(
    const __bf16* __restrict__ qkv, const __bf16* __restrict__ qkvc,
    __bf16* __restrict__ y) {
  const int br = blockIdx.x, h = blockIdx.y, b = blockIdx.z;
  const int tid = threadIdx.x, lane = tid & 31, wave = tid >> 5;
  const int lrow = lane & 15, lhalf = lane >> 4;

  __shared__ __bf16 Ksh[32 * 64];       // [key][d]
  __shared__ __bf16 Vtsh[64 * 32];      // [d][key]
  __shared__ __bf16 Psh[16][16 * 32];   // per-wave P tile [row][key]
  __shared__ float kcs[kD], vcs[kD];

  if (tid < kD) {
    kcs[tid] = (float)qkvc[(size_t)b * kC3 + kC + h * kD + tid];
    vcs[tid] = (float)qkvc[(size_t)b * kC3 + 2 * kC + h * kD + tid];
  }
  __syncthreads();

  // Load Q fragments (rows = this wave's 16 query tokens), two d-chunks of 32.
  const __bf16* qbase =
      qkv + ((size_t)b * kN + 1 + (size_t)br * kSPB + wave * 16 + lrow) * kC3 + h * kD;
  v16bf qf[2];
#pragma unroll
  for (int c = 0; c < 2; ++c)
    qf[c] = make_frag(qbase + c * 32 + lhalf * 8, qbase + c * 32 + 16 + lhalf * 8);

  // --- cls key as initial flash state ---
  float pd = 0.f;
#pragma unroll
  for (int c = 0; c < 2; ++c)
#pragma unroll
    for (int j = 0; j < 8; ++j) {
      pd += (float)qf[c][j] * kcs[c * 32 + lhalf * 8 + j];
      pd += (float)qf[c][j + 8] * kcs[c * 32 + 16 + lhalf * 8 + j];
    }
  pd += __shfl_xor(pd, 16);          // full dot for row (lane&15) in both halves
  const float scls = pd * kScale;

  float m[8], l[8], alpha[8];
  v8f o[4];
#pragma unroll
  for (int v = 0; v < 8; ++v) {
    m[v] = __shfl(scls, lhalf * 8 + v);  // s_cls for this C-layout row
    l[v] = 1.0f;                          // p_cls = exp(0) = 1
  }
#pragma unroll
  for (int nt = 0; nt < 4; ++nt) {
    const float vcv = vcs[nt * 16 + lrow];
#pragma unroll
    for (int v = 0; v < 8; ++v) o[nt][v] = vcv;  // O = p_cls * vc
  }

  // staging roles: each thread moves 4 bf16 (8B) of K and of V per block
  const int e = tid * 4;
  const int kk = e >> 6;   // key 0..31
  const int dd = e & 63;   // dim (multiple of 4)

  // --- 8 key-blocks of 32 keys (tokens 1+br*256+kt ...) ---
  for (int kt = 0; kt < kSPB; kt += 32) {
    const size_t rowoff =
        ((size_t)b * kN + 1 + (size_t)br * kSPB + kt + kk) * kC3 + h * kD + dd;
    // K: identical layout global->LDS => async DMA path (ASYNCcnt)
    async_copy_b64(&Ksh[kk * 64 + dd], qkv + rowoff + kC);
    // V: needs transpose, stage through VGPRs
    v4bf vv = *(const v4bf*)(qkv + rowoff + 2 * kC);
#pragma unroll
    for (int j = 0; j < 4; ++j) Vtsh[(dd + j) * 32 + kk] = vv[j];
    wait_asynccnt0();
    __syncthreads();

    // S = Q K^T for two 16-key sub-tiles (WMMA K-dim = head dim, 2 chunks)
    v8f st[2];
#pragma unroll
    for (int t = 0; t < 2; ++t) {
      v8f sacc;
#pragma unroll
      for (int v = 0; v < 8; ++v) sacc[v] = 0.f;
#pragma unroll
      for (int c = 0; c < 2; ++c) {
        const __bf16* bp = &Ksh[(t * 16 + lrow) * 64 + c * 32 + lhalf * 8];
        sacc = wmma_bf16(qf[c], make_frag(bp, bp + 16), sacc);
      }
      st[t] = sacc;
    }

    // Online softmax update over these 32 keys; write P (bf16) to LDS.
#pragma unroll
    for (int v = 0; v < 8; ++v) {
      float sa = st[0][v] * kScale;
      float sb = st[1][v] * kScale;
      float mx = fmaxf(sa, sb);
#pragma unroll
      for (int d2 = 8; d2 >= 1; d2 >>= 1) mx = fmaxf(mx, __shfl_xor(mx, d2));
      const float mnew = fmaxf(m[v], mx);
      alpha[v] = __expf(m[v] - mnew);
      const float pa = __expf(sa - mnew);
      const float pb = __expf(sb - mnew);
      float ls = pa + pb;
#pragma unroll
      for (int d2 = 8; d2 >= 1; d2 >>= 1) ls += __shfl_xor(ls, d2);
      l[v] = l[v] * alpha[v] + ls;
      m[v] = mnew;
      const int pr = (lhalf * 8 + v) * 32;
      Psh[wave][pr + lrow] = (__bf16)pa;
      Psh[wave][pr + 16 + lrow] = (__bf16)pb;
    }

    // O = O*alpha + P V  (A-frag from Psh, B-frags from transposed V)
    const __bf16* pp = &Psh[wave][lrow * 32];
    v16bf pf = make_frag(pp + lhalf * 8, pp + 16 + lhalf * 8);
#pragma unroll
    for (int nt = 0; nt < 4; ++nt) {
#pragma unroll
      for (int v = 0; v < 8; ++v) o[nt][v] *= alpha[v];
      const __bf16* vp = &Vtsh[(nt * 16 + lrow) * 32 + lhalf * 8];
      o[nt] = wmma_bf16(pf, make_frag(vp, vp + 16), o[nt]);
    }
    __syncthreads();
  }

  // Epilogue: normalize and store this wave's 16x64 output tile.
  const size_t ybase =
      ((size_t)b * kN + 1 + (size_t)br * kSPB + wave * 16) * kC + h * kD;
#pragma unroll
  for (int nt = 0; nt < 4; ++nt)
#pragma unroll
    for (int v = 0; v < 8; ++v) {
      const int r = lhalf * 8 + v;
      y[ybase + (size_t)r * kC + nt * 16 + lrow] = (__bf16)(o[nt][v] / l[v]);
    }
}

// ---------------------------------------------------------------------------
extern "C" void kernel_launch(void* const* d_in, const int* in_sizes, int n_in,
                              void* d_out, int out_size, void* d_ws, size_t ws_size,
                              hipStream_t stream) {
  (void)in_sizes; (void)n_in; (void)out_size; (void)ws_size;
  const float* x      = (const float*)d_in[0];
  const float* W_qkv  = (const float*)d_in[1];
  const float* W_proj = (const float*)d_in[2];
  const float* b_proj = (const float*)d_in[3];
  float* out = (float*)d_out;

  char* ws = (char*)d_ws;
  size_t off = 0;
  auto take = [&](size_t bytes) -> char* {
    char* p = ws + off;
    off += (bytes + 255) & ~(size_t)255;
    return p;
  };
  __bf16* qkv_bf   = (__bf16*)take((size_t)kM * kC3 * 2);  // 75.6 MB
  __bf16* x_bf     = (__bf16*)take((size_t)kM * kC * 2);   // 25.2 MB
  __bf16* y_bf     = (__bf16*)take((size_t)kM * kC * 2);   // 25.2 MB
  __bf16* Wqkv_bf  = (__bf16*)take((size_t)kC * kC3 * 2);
  __bf16* Wproj_bf = (__bf16*)take((size_t)kC * kC * 2);
  float*  cls_f    = (float*)take((size_t)kB * kC * 4);
  __bf16* qkvc_bf  = (__bf16*)take((size_t)kB * kC3 * 2);

  {
    long n = (long)kC * kC3;
    cvt_f32_bf16_kernel<<<(n + 255) / 256, 256, 0, stream>>>(W_qkv, Wqkv_bf, n);
  }
  {
    long n = (long)kC * kC;
    cvt_f32_bf16_kernel<<<(n + 255) / 256, 256, 0, stream>>>(W_proj, Wproj_bf, n);
  }
  {
    long n = (long)kM * kC;
    cvt_f32_bf16_kernel<<<(n + 255) / 256, 256, 0, stream>>>(x, x_bf, n);
  }

  // qkv = x @ W_qkv  (bf16 out)
  gemm_wmma_kernel<true><<<dim3((kM + 127) / 128, kC3 / 128), 256, 0, stream>>>(
      x_bf, Wqkv_bf, nullptr, qkv_bf, kM, kC3, kC);

  // cls token global attention + residual
  cls_attn_kernel<<<kB * kH, 256, 0, stream>>>(qkv_bf, x, cls_f, y_bf);

  // qkv_c = cls @ W_qkv
  qkvc_kernel<<<dim3(kB, kC3 / 256), 256, 0, stream>>>(cls_f, Wqkv_bf, qkvc_bf);

  // per-branch windowed attention (cls as extra key/value)
  branch_attn_kernel<<<dim3(kNB, kH, kB), 512, 0, stream>>>(qkv_bf, qkvc_bf, y_bf);

  // out = y @ W_proj + b_proj  (f32 out)
  gemm_wmma_kernel<false><<<dim3((kM + 127) / 128, kC / 128), 256, 0, stream>>>(
      y_bf, Wproj_bf, b_proj, (void*)out, kM, kC, kC);
}
